// BiLSTMCRF_43628277793533
// MI455X (gfx1250) — compile-verified, matrix operand order checked by reading, not run on hardware
//
#include <hip/hip_runtime.h>
#include <hip/hip_bf16.h>
#include <stdint.h>

// ---------------- CDNA5 WMMA types ----------------
typedef __attribute__((ext_vector_type(16))) __bf16 v16bf;
typedef __attribute__((ext_vector_type(8)))  float  v8f;
typedef __attribute__((ext_vector_type(2)))  float  v2f;

union FragB16 { v16bf v; uint4 q[2]; };   // 32 bytes: one 16x32 bf16 A/B fragment per lane

// ---------------- problem constants ----------------
#define SEQ_T  2048
#define BATCH  32
#define INDIM  512
#define EMB    256
#define HIDDEN 256
#define HH     128
#define G4     512     // 4*HH gate width
#define NL     16      // labels
#define TBROWS (SEQ_T*BATCH)   // 65536

__device__ __forceinline__ unsigned short f2bf(float x) {
  unsigned u = __float_as_uint(x);
  unsigned r = 0x7FFFu + ((u >> 16) & 1u);     // round-to-nearest-even
  return (unsigned short)((u + r) >> 16);
}
__device__ __forceinline__ float sigm(float x) { return 1.0f / (1.0f + __expf(-x)); }

__device__ __forceinline__ v8f v8zero() {
  v8f z;
  #pragma unroll
  for (int e = 0; e < 8; ++e) z[e] = 0.0f;
  return z;
}

// ---------------- prep kernels ----------------
__global__ void cvt_f32_bf16(const float* __restrict__ in, unsigned short* __restrict__ out, int n) {
  int i = blockIdx.x * blockDim.x + threadIdx.x;
  if (i < n) out[i] = f2bf(in[i]);
}

// Wfused[j][k] = sum_e Wih[j][e] * inW[e][k]   (512x512, bf16 out)
__global__ void fuse_w(const float* __restrict__ Wih, const float* __restrict__ inW,
                       unsigned short* __restrict__ out) {
  int idx = blockIdx.x * blockDim.x + threadIdx.x;   // j*512 + k
  int j = idx >> 9, k = idx & 511;
  float s = 0.0f;
  for (int e = 0; e < EMB; ++e) s += Wih[j * EMB + e] * inW[e * INDIM + k];
  out[idx] = f2bf(s);
}

// bias[j] = b[j] + sum_e Wih[j][e]*in_b[e]
__global__ void fuse_bias(const float* __restrict__ Wih, const float* __restrict__ inb,
                          const float* __restrict__ bg, float* __restrict__ out) {
  int j = blockIdx.x * blockDim.x + threadIdx.x;
  if (j < G4) {
    float s = bg[j];
    for (int e = 0; e < EMB; ++e) s += Wih[j * EMB + e] * inb[e];
    out[j] = s;
  }
}

// ---------------- big pre-gate GEMM: C[65536][512] = A[65536][512]bf16 @ W[512][512]^T bf16 + bias
// grid = (M/128, 8 N-groups); 256 threads = 8 waves; wave owns one 16-row M tile x 4 N tiles.
// K loop is software-pipelined: fragments for chunk kc+1 are in flight while chunk kc's
// WMMAs issue, so waits are partial instead of load-latency barriers.
__global__ void gate_gemm(const unsigned short* __restrict__ A,
                          const unsigned short* __restrict__ W,
                          const float* __restrict__ bias,
                          float* __restrict__ C) {
  const int K = INDIM;
  const int NKC = K / 32;                    // 16 K chunks
  int wave = threadIdx.x >> 5, lane = threadIdx.x & 31;
  int half = lane >> 4, r = lane & 15;
  int row0 = blockIdx.x * 128 + wave * 16;
  int ng = blockIdx.y;                       // N group: 64 columns
  const unsigned short* Arow = A + (size_t)(row0 + r) * K;
  const unsigned short* Wbase = W + (size_t)(ng * 64 + r) * K;   // + j*16*K per tile

  v8f acc[4];
  #pragma unroll
  for (int j = 0; j < 4; ++j) acc[j] = v8zero();

  FragB16 a[2];
  FragB16 b[2][4];

  // prologue: stage 0
  {
    int kb = half * 8;
    a[0].q[0] = *(const uint4*)(Arow + kb);
    a[0].q[1] = *(const uint4*)(Arow + kb + 16);
    #pragma unroll
    for (int j = 0; j < 4; ++j) {
      const unsigned short* Wrow = Wbase + (size_t)j * 16 * K + kb;
      b[0][j].q[0] = *(const uint4*)(Wrow);
      b[0][j].q[1] = *(const uint4*)(Wrow + 16);
    }
  }

  #pragma unroll
  for (int kc = 0; kc < NKC; ++kc) {
    const int cur = kc & 1, nxt = cur ^ 1;
    if (kc + 1 < NKC) {                      // prefetch next chunk
      int kb = (kc + 1) * 32 + half * 8;
      a[nxt].q[0] = *(const uint4*)(Arow + kb);
      a[nxt].q[1] = *(const uint4*)(Arow + kb + 16);
      #pragma unroll
      for (int j = 0; j < 4; ++j) {
        const unsigned short* Wrow = Wbase + (size_t)j * 16 * K + kb;
        b[nxt][j].q[0] = *(const uint4*)(Wrow);
        b[nxt][j].q[1] = *(const uint4*)(Wrow + 16);
      }
    }
    #pragma unroll
    for (int j = 0; j < 4; ++j)
      acc[j] = __builtin_amdgcn_wmma_f32_16x16x32_bf16(
          false, a[cur].v, false, b[cur][j].v, (short)0, acc[j], false, false);
  }

  #pragma unroll
  for (int j = 0; j < 4; ++j) {
    int n = ng * 64 + j * 16 + r;
    float bs = bias[n];
    #pragma unroll
    for (int e = 0; e < 8; ++e)
      C[(size_t)(row0 + half * 8 + e) * G4 + n] = acc[j][e] + bs;
  }
}

// ---------------- sequential BiLSTM recurrence ----------------
// grid = 2 (direction), block = 1024 threads = 32 waves.
// Per step: g(32x512) = h(32x128) @ Whh^T via WMMA bf16; Whh resident in VGPRs
// (2 tiles x 4 K-chunks = 64 VGPRs/wave); f32 cell state in registers (4/thread).
__global__ void lstm_kernel(const unsigned short* __restrict__ WhhAll,
                            const float* __restrict__ GpreAll,
                            const float* __restrict__ h0,
                            const float* __restrict__ c0,
                            float* __restrict__ hs) {
  extern __shared__ char smem[];
  float* g = (float*)smem;                                          // 32*512 f32 = 64KB
  unsigned short* hb = (unsigned short*)(smem + 32 * G4 * sizeof(float)); // 32*128 bf16 = 8KB

  const int dir = blockIdx.x;
  const unsigned short* Whh = WhhAll + (size_t)dir * G4 * HH;
  const float* Gpre = GpreAll + (size_t)dir * TBROWS * G4;

  int tid = threadIdx.x;
  int wave = tid >> 5, lane = tid & 31, half = lane >> 4, r = lane & 15;
  int mtile = wave & 1;           // which 16-row batch tile
  int ngroup = wave >> 1;         // 0..15, covers 2 N tiles each

  // preload Whh B fragments (stay in VGPRs for all 2048 steps)
  v16bf Bfr[2][4];
  #pragma unroll
  for (int j = 0; j < 2; ++j) {
    int n0 = (ngroup * 2 + j) * 16;
    #pragma unroll
    for (int kc = 0; kc < 4; ++kc) {
      const unsigned short* p = Whh + (size_t)(n0 + r) * HH + kc * 32 + half * 8;
      FragB16 f;
      f.q[0] = *(const uint4*)(p);
      f.q[1] = *(const uint4*)(p + 16);
      Bfr[j][kc] = f.v;
    }
  }

  // persistent cell state: thread owns 4 consecutive (b,kk) elements
  int base = tid * 4;
  int b = base >> 7, kk = base & 127;
  float cReg[4];
  #pragma unroll
  for (int e = 0; e < 4; ++e) {
    cReg[e] = c0[(dir * BATCH + b) * HH + kk + e];
    hb[b * HH + kk + e] = f2bf(h0[(dir * BATCH + b) * HH + kk + e]);
  }
  __syncthreads();

  for (int t = 0; t < SEQ_T; ++t) {
    int tt = dir ? (SEQ_T - 1 - t) : t;

    // --- matmul phase: g = h @ Whh^T
    v8f acc[2];
    #pragma unroll
    for (int j = 0; j < 2; ++j) acc[j] = v8zero();
    #pragma unroll
    for (int kc = 0; kc < 4; ++kc) {
      int ab = (mtile * 16 + r) * HH + kc * 32 + half * 8;
      FragB16 a;
      a.q[0] = *(const uint4*)(hb + ab);
      a.q[1] = *(const uint4*)(hb + ab + 16);
      #pragma unroll
      for (int j = 0; j < 2; ++j)
        acc[j] = __builtin_amdgcn_wmma_f32_16x16x32_bf16(
            false, a.v, false, Bfr[j][kc], (short)0, acc[j], false, false);
    }
    #pragma unroll
    for (int j = 0; j < 2; ++j) {
      int n0 = (ngroup * 2 + j) * 16;
      #pragma unroll
      for (int e = 0; e < 8; ++e)
        g[(mtile * 16 + half * 8 + e) * G4 + n0 + r] = acc[j][e];
    }
    __syncthreads();

    // --- elementwise LSTM cell (gate order i, f, g, o)
    const float* Gt = Gpre + (size_t)(tt * BATCH + b) * G4 + kk;
    float* hrow = hs + (size_t)(tt * BATCH + b) * HIDDEN + dir * HH;
    #pragma unroll
    for (int e = 0; e < 4; ++e) {
      int col = kk + e;
      float gi = g[b * G4 + col]            + Gt[e];
      float gf = g[b * G4 + 128 + col]      + Gt[128 + e];
      float gg = g[b * G4 + 256 + col]      + Gt[256 + e];
      float go = g[b * G4 + 384 + col]      + Gt[384 + e];
      float c = sigm(gf) * cReg[e] + sigm(gi) * tanhf(gg);
      cReg[e] = c;
      float h = sigm(go) * tanhf(c);
      hrow[col] = h;
      hb[b * HH + col] = f2bf(h);
    }
    __syncthreads();
  }
}

// ---------------- output projection: feats[65536][16] = hs @ outW^T + outb, f32 WMMA
__global__ void outproj(const float* __restrict__ hsrc, const float* __restrict__ outW,
                        const float* __restrict__ outb, float* __restrict__ feats) {
  int wave = threadIdx.x >> 5, lane = threadIdx.x & 31;
  int half = lane >> 4, r = lane & 15;
  int row0 = blockIdx.x * 128 + wave * 16;
  const float* Arow = hsrc + (size_t)(row0 + r) * HIDDEN;
  const float* Brow = outW + r * HIDDEN;            // out_W[r][*], B[k][n]=outW[n][k]
  v8f acc = v8zero();
  for (int kc = 0; kc < HIDDEN / 4; ++kc) {
    int ks = kc * 4 + half * 2;
    v2f a, b;
    a[0] = Arow[ks]; a[1] = Arow[ks + 1];
    b[0] = Brow[ks]; b[1] = Brow[ks + 1];
    acc = __builtin_amdgcn_wmma_f32_16x16x4_f32(
        false, a, false, b, (short)0, acc, false, false);
  }
  float bias = outb[r];
  #pragma unroll
  for (int e = 0; e < 8; ++e)
    feats[(size_t)(row0 + half * 8 + e) * NL + r] = acc[e] + bias;
}

// ---------------- Viterbi decode: 1 block, 512 threads = (batch 32) x (labels 16)
__global__ void viterbi_kernel(const float* __restrict__ feats, const float* __restrict__ trans,
                               int* __restrict__ bp, float* __restrict__ outScore,
                               int* __restrict__ outPath) {
  __shared__ float fv[BATCH][NL];
  __shared__ float tr[NL][NL];
  __shared__ float tbuf[BATCH][NL];
  __shared__ int last[BATCH];
  int tid = threadIdx.x;
  int b = tid >> 4, l = tid & 15;
  if (tid < NL * NL) ((float*)tr)[tid] = trans[tid];
  fv[b][l] = (l == 0) ? 0.0f : -10000.0f;        // START = 0
  __syncthreads();

  for (int t = 0; t < SEQ_T; ++t) {
    float best = -3.4e38f; int arg = 0;
    #pragma unroll
    for (int pl = 0; pl < NL; ++pl) {
      float s = fv[b][pl] + tr[l][pl];
      if (s > best) { best = s; arg = pl; }
    }
    float nf = best + feats[(size_t)(t * BATCH + b) * NL + l];
    bp[(size_t)(t * BATCH + b) * NL + l] = arg;
    __syncthreads();
    fv[b][l] = nf;
    __syncthreads();
  }

  tbuf[b][l] = fv[b][l] + tr[1][l];              // STOP = 1
  __syncthreads();
  if (l == 0) {
    float best = tbuf[b][0]; int arg = 0;
    for (int pl = 1; pl < NL; ++pl)
      if (tbuf[b][pl] > best) { best = tbuf[b][pl]; arg = pl; }
    outScore[b] = best;
    last[b] = arg;
  }
  __syncthreads();
  if (tid < BATCH) {
    int lbl = last[tid];
    for (int t = SEQ_T - 1; t >= 0; --t) {
      outPath[t * BATCH + tid] = lbl;
      lbl = bp[(size_t)(t * BATCH + tid) * NL + lbl];
    }
  }
}

// ---------------- host launch ----------------
extern "C" void kernel_launch(void* const* d_in, const int* in_sizes, int n_in,
                              void* d_out, int out_size, void* d_ws, size_t ws_size,
                              hipStream_t stream) {
  const float* src  = (const float*)d_in[0];
  const float* inW  = (const float*)d_in[1];
  const float* inb  = (const float*)d_in[2];
  const float* WihF = (const float*)d_in[3];
  const float* WhhF = (const float*)d_in[4];
  const float* bF   = (const float*)d_in[5];
  const float* WihB = (const float*)d_in[6];
  const float* WhhB = (const float*)d_in[7];
  const float* bB   = (const float*)d_in[8];
  const float* outW = (const float*)d_in[9];
  const float* outb = (const float*)d_in[10];
  const float* trn  = (const float*)d_in[11];
  const float* h0   = (const float*)d_in[12];
  const float* c0   = (const float*)d_in[13];

  char* ws = (char*)d_ws;
  size_t off = 0;
  auto alloc = [&](size_t bytes) { size_t p = off; off += (bytes + 255) & ~(size_t)255; return p; };
  unsigned short* srcbf = (unsigned short*)(ws + alloc((size_t)TBROWS * INDIM * 2));
  unsigned short* wfuse = (unsigned short*)(ws + alloc((size_t)2 * G4 * INDIM * 2));
  float*          biasf = (float*)        (ws + alloc((size_t)2 * G4 * 4));
  unsigned short* whhbf = (unsigned short*)(ws + alloc((size_t)2 * G4 * HH * 2));
  float*          gpre  = (float*)        (ws + alloc((size_t)2 * TBROWS * G4 * 4));
  float*          hsbuf = (float*)        (ws + alloc((size_t)TBROWS * HIDDEN * 4));
  float*          feats = (float*)        (ws + alloc((size_t)TBROWS * NL * 4));
  int*            bpbuf = (int*)          (ws + alloc((size_t)TBROWS * NL * 4));
  (void)ws_size; (void)in_sizes; (void)n_in; (void)out_size;

  // prep: bf16 conversions + algebraic fusion (Wih @ in_W), fused biases
  cvt_f32_bf16<<<(TBROWS * INDIM) / 256, 256, 0, stream>>>(src, srcbf, TBROWS * INDIM);
  cvt_f32_bf16<<<(G4 * HH) / 256, 256, 0, stream>>>(WhhF, whhbf, G4 * HH);
  cvt_f32_bf16<<<(G4 * HH) / 256, 256, 0, stream>>>(WhhB, whhbf + G4 * HH, G4 * HH);
  fuse_w<<<(G4 * INDIM) / 256, 256, 0, stream>>>(WihF, inW, wfuse);
  fuse_w<<<(G4 * INDIM) / 256, 256, 0, stream>>>(WihB, inW, wfuse + (size_t)G4 * INDIM);
  fuse_bias<<<2, 256, 0, stream>>>(WihF, inb, bF, biasf);
  fuse_bias<<<2, 256, 0, stream>>>(WihB, inb, bB, biasf + G4);

  // time-parallel pre-gate GEMMs (bf16 WMMA), N-groups in grid.y, pipelined K loop
  dim3 ggrid(TBROWS / 128, 8);
  gate_gemm<<<ggrid, 256, 0, stream>>>(srcbf, wfuse, biasf, gpre);
  gate_gemm<<<ggrid, 256, 0, stream>>>(srcbf, wfuse + (size_t)G4 * INDIM,
                                       biasf + G4, gpre + (size_t)TBROWS * G4);

  // sequential recurrence, one workgroup per direction (32 waves each)
  size_t lds = (size_t)32 * G4 * 4 + (size_t)32 * HH * 2;   // 72KB dynamic LDS
  lstm_kernel<<<2, 1024, lds, stream>>>(whhbf, gpre, h0, c0, hsbuf);

  // output projection (f32 WMMA) + Viterbi
  outproj<<<TBROWS / 128, 256, 0, stream>>>(hsbuf, outW, outb, feats);
  viterbi_kernel<<<1, 512, 0, stream>>>(feats, trn, bpbuf,
                                        (float*)d_out, (int*)d_out + BATCH);
}
